// GraphConvolution_73796128080687
// MI455X (gfx1250) — matrix operand order repaired and checked by reading, not compiled, not run on hardware
//
#include <hip/hip_runtime.h>
#include <hip/hip_bf16.h>

typedef __attribute__((ext_vector_type(16))) _Float16 v16h;
typedef __attribute__((ext_vector_type(8)))  _Float16 v8h;
typedef __attribute__((ext_vector_type(8)))  float    v8f;

#define D_FEAT 128

// ---------------- conversion kernels ----------------

__global__ void convert_x_f16(const float* __restrict__ x, _Float16* __restrict__ xh, int n) {
    int i = blockIdx.x * blockDim.x + threadIdx.x;
    if (i < n) xh[i] = (_Float16)x[i];
}

// transpose + convert both weight matrices: wt[n*128+k] = (f16) W[k*128+n]
__global__ void convert_w_f16t(const float* __restrict__ W0, const float* __restrict__ W1,
                               _Float16* __restrict__ w0t, _Float16* __restrict__ w1t) {
    int i = blockIdx.x * blockDim.x + threadIdx.x; // 0 .. 16383
    if (i < D_FEAT * D_FEAT) {
        int k = i >> 7, n = i & 127;
        w0t[n * D_FEAT + k] = (_Float16)W0[i];
        w1t[n * D_FEAT + k] = (_Float16)W1[i];
    }
}

// ---------------- CSR build ----------------

__global__ void zero_counts(int* __restrict__ c0, int* __restrict__ c1, int n) {
    int i = blockIdx.x * blockDim.x + threadIdx.x;
    if (i < n) { c0[i] = 0; c1[i] = 0; }
}

__global__ void count_edges(const int* __restrict__ ei0, const int* __restrict__ ei1,
                            int* __restrict__ c0, int* __restrict__ c1, int E) {
    int e = blockIdx.x * blockDim.x + threadIdx.x;
    if (e < E) {
        atomicAdd(&c0[ei0[e] + 1], 1);   // ei[0..E) = dst row
        atomicAdd(&c1[ei1[e] + 1], 1);
    }
}

// single-block inclusive scan (in place), n up to ~50k: chunked Hillis-Steele
__global__ void __launch_bounds__(1024) scan_inclusive(int* __restrict__ a, int n) {
    __shared__ int buf[1024];
    __shared__ int carry;
    int tid = threadIdx.x;
    if (tid == 0) carry = 0;
    __syncthreads();
    for (int base = 0; base < n; base += 1024) {
        int i = base + tid;
        int v = (i < n) ? a[i] : 0;
        buf[tid] = v;
        __syncthreads();
        for (int off = 1; off < 1024; off <<= 1) {
            int t = (tid >= off) ? buf[tid - off] : 0;
            __syncthreads();
            buf[tid] += t;
            __syncthreads();
        }
        int res = buf[tid] + carry;
        if (i < n) a[i] = res;
        __syncthreads();                 // everyone done reading old carry
        if (tid == 1023) carry = res;
        __syncthreads();
    }
}

__global__ void copy_cursor(const int* __restrict__ o0, const int* __restrict__ o1,
                            int* __restrict__ cu0, int* __restrict__ cu1, int n) {
    int i = blockIdx.x * blockDim.x + threadIdx.x;
    if (i < n) { cu0[i] = o0[i]; cu1[i] = o1[i]; }
}

__global__ void fill_perm(const int* __restrict__ ei0, const int* __restrict__ ei1,
                          int* __restrict__ cu0, int* __restrict__ cu1,
                          int* __restrict__ p0, int* __restrict__ p1, int E) {
    int e = blockIdx.x * blockDim.x + threadIdx.x;
    if (e < E) {
        int pos0 = atomicAdd(&cu0[ei0[e]], 1);
        p0[pos0] = e;
        int pos1 = atomicAdd(&cu1[ei1[e]], 1);
        p1[pos1] = e;
    }
}

// ---------------- WMMA GEMM: pre0 = X@W0, pre1 = X@W1 (f16 in, f32 acc) ----------------
// Block = 256 threads = 8 waves; block b handles M-tile b (16 rows), wave w handles N-tile w.
// Each wave loads A once per K-chunk and feeds two accumulators (W0 and W1).

__global__ void __launch_bounds__(256) gemm_dual_wmma(
        const _Float16* __restrict__ xh,
        const _Float16* __restrict__ w0t,   // [n][k] transposed, f16
        const _Float16* __restrict__ w1t,
        float* __restrict__ pre0, float* __restrict__ pre1) {
    const int lane  = threadIdx.x & 31;
    const int ntile = threadIdx.x >> 5;      // 0..7
    const int M0    = blockIdx.x * 16;
    const int half  = lane >> 4;             // 0|1
    const int l15   = lane & 15;             // A: row M ; B/C: col N

    const _Float16* xrow = xh + (size_t)(M0 + l15) * D_FEAT;
    const _Float16* b0p  = w0t + (size_t)(ntile * 16 + l15) * D_FEAT;
    const _Float16* b1p  = w1t + (size_t)(ntile * 16 + l15) * D_FEAT;

    v8f c0 = {};
    v8f c1 = {};
#pragma unroll
    for (int kc = 0; kc < 4; ++kc) {
        const int kb = kc * 32;
        // A 16x32 f16: element e -> K = half*16 + e
        v16h a = *(const v16h*)(xrow + kb + half * 16);
        // B 32x16 f16: lane holds column N, element e -> K = half*16 + e (contiguous in wt)
        v16h b0 = *(const v16h*)(b0p + kb + half * 16);
        v16h b1 = *(const v16h*)(b1p + kb + half * 16);
        c0 = __builtin_amdgcn_wmma_f32_16x16x32_f16(false, a, false, b0, (short)0, c0, false, false);
        c1 = __builtin_amdgcn_wmma_f32_16x16x32_f16(false, a, false, b1, (short)0, c1, false, false);
    }
    // C/D layout: VGPR r, lane -> row M0 + 8*half + r, col ntile*16 + l15
    const int col = ntile * 16 + l15;
    float* o0 = pre0 + (size_t)(M0 + 8 * half) * D_FEAT + col;
    float* o1 = pre1 + (size_t)(M0 + 8 * half) * D_FEAT + col;
#pragma unroll
    for (int r = 0; r < 8; ++r) {
        o0[(size_t)r * D_FEAT] = c0[r];
        o1[(size_t)r * D_FEAT] = c1[r];
    }
}

// ---------------- fused SpMM (gather form) + bias + relu ----------------
// one wave per destination node, 8 nodes per 256-thread block.
// lane handles 4 consecutive features as float4 -> global_load_b128 gathers.
// perm/ei/vals addresses are wave-uniform -> scalar loads.
// pre0/pre1 (51MB total) are resident in the 192MB L2, so the ~820MB of
// random row gathers are L2 hits rather than HBM random access.

__global__ void __launch_bounds__(256) aggregate_relu(
        const float* __restrict__ pre0, const float* __restrict__ pre1,
        const float* __restrict__ bias,
        const float* __restrict__ vals0, const float* __restrict__ vals1,
        const int* __restrict__ ei0, const int* __restrict__ ei1,
        const int* __restrict__ off0, const int* __restrict__ off1,
        const int* __restrict__ perm0, const int* __restrict__ perm1,
        float* __restrict__ out, int E, int N) {
    const int wave = threadIdx.x >> 5;
    const int lane = threadIdx.x & 31;
    const int n = blockIdx.x * 8 + wave;
    if (n >= N) return;

    const float4* b4 = (const float4*)bias;
    float4 acc = b4[lane];

    const float4* p0 = (const float4*)pre0;   // 32 float4 per row
    const float4* p1 = (const float4*)pre1;

    int s = off0[n], e = off0[n + 1];
    for (int i = s; i < e; ++i) {
        int eid = perm0[i];
        int src = ei0[E + eid];                // row 1 of edge_index = src
        float v = vals0[eid];
        if (i + 1 < e) {                       // prefetch next gathered row
            int nsrc = ei0[E + perm0[i + 1]];
            __builtin_prefetch(&p0[(size_t)nsrc * 32 + lane], 0, 1);
        }
        float4 r = p0[(size_t)src * 32 + lane];
        acc.x += v * r.x; acc.y += v * r.y; acc.z += v * r.z; acc.w += v * r.w;
    }
    s = off1[n]; e = off1[n + 1];
    for (int i = s; i < e; ++i) {
        int eid = perm1[i];
        int src = ei1[E + eid];
        float v = vals1[eid];
        if (i + 1 < e) {
            int nsrc = ei1[E + perm1[i + 1]];
            __builtin_prefetch(&p1[(size_t)nsrc * 32 + lane], 0, 1);
        }
        float4 r = p1[(size_t)src * 32 + lane];
        acc.x += v * r.x; acc.y += v * r.y; acc.z += v * r.z; acc.w += v * r.w;
    }
    float4 res;
    res.x = fmaxf(acc.x, 0.0f);
    res.y = fmaxf(acc.y, 0.0f);
    res.z = fmaxf(acc.z, 0.0f);
    res.w = fmaxf(acc.w, 0.0f);
    ((float4*)out)[(size_t)n * 32 + lane] = res;
}

// ---------------- launcher ----------------

extern "C" void kernel_launch(void* const* d_in, const int* in_sizes, int n_in,
                              void* d_out, int out_size, void* d_ws, size_t ws_size,
                              hipStream_t stream) {
    const float* x     = (const float*)d_in[0];
    const float* W0    = (const float*)d_in[1];
    const float* W1    = (const float*)d_in[2];
    const float* bias  = (const float*)d_in[3];
    const float* vals0 = (const float*)d_in[4];
    const float* vals1 = (const float*)d_in[5];
    const int*   ei0   = (const int*)d_in[6];
    const int*   ei1   = (const int*)d_in[7];
    float*       out   = (float*)d_out;

    const int N = in_sizes[0] / D_FEAT;   // 50000
    const int E = in_sizes[4];            // 800000

    // workspace layout (256B-aligned slabs)
    char* ws = (char*)d_ws;
    size_t off = 0;
    auto alloc = [&](size_t bytes) -> char* {
        char* p = ws + off;
        off = (off + bytes + 255) & ~(size_t)255;
        return p;
    };
    _Float16* xh   = (_Float16*)alloc((size_t)N * D_FEAT * sizeof(_Float16));
    _Float16* w0t  = (_Float16*)alloc((size_t)D_FEAT * D_FEAT * sizeof(_Float16));
    _Float16* w1t  = (_Float16*)alloc((size_t)D_FEAT * D_FEAT * sizeof(_Float16));
    float*    pre0 = (float*)alloc((size_t)N * D_FEAT * sizeof(float));
    float*    pre1 = (float*)alloc((size_t)N * D_FEAT * sizeof(float));
    int*      off0 = (int*)alloc((size_t)(N + 1) * sizeof(int));
    int*      off1 = (int*)alloc((size_t)(N + 1) * sizeof(int));
    int*      cur0 = (int*)alloc((size_t)N * sizeof(int));
    int*      cur1 = (int*)alloc((size_t)N * sizeof(int));
    int*      prm0 = (int*)alloc((size_t)E * sizeof(int));
    int*      prm1 = (int*)alloc((size_t)E * sizeof(int));
    (void)ws_size; (void)n_in; (void)out_size;

    const int T = 256;
    // 1) CSR build
    zero_counts<<<(N + 1 + T - 1) / T, T, 0, stream>>>(off0, off1, N + 1);
    count_edges<<<(E + T - 1) / T, T, 0, stream>>>(ei0, ei1, off0, off1, E);
    scan_inclusive<<<1, 1024, 0, stream>>>(off0, N + 1);
    scan_inclusive<<<1, 1024, 0, stream>>>(off1, N + 1);
    copy_cursor<<<(N + T - 1) / T, T, 0, stream>>>(off0, off1, cur0, cur1, N);
    fill_perm<<<(E + T - 1) / T, T, 0, stream>>>(ei0, ei1, cur0, cur1, prm0, prm1, E);

    // 2) f16 conversion + dual WMMA GEMM
    convert_x_f16<<<((size_t)N * D_FEAT + T - 1) / T, T, 0, stream>>>(x, xh, N * D_FEAT);
    convert_w_f16t<<<(D_FEAT * D_FEAT + T - 1) / T, T, 0, stream>>>(W0, W1, w0t, w1t);
    gemm_dual_wmma<<<N / 16, 256, 0, stream>>>(xh, w0t, w1t, pre0, pre1);

    // 3) fused gather-SpMM + bias + relu (wave per node, float4 lanes)
    aggregate_relu<<<(N + 7) / 8, 256, 0, stream>>>(pre0, pre1, bias, vals0, vals1,
                                                    ei0, ei1, off0, off1, prm0, prm1,
                                                    out, E, N);
}
